// attention_42331197669711
// MI455X (gfx1250) — matrix operand order repaired and checked by reading, not compiled, not run on hardware
//
#include <hip/hip_runtime.h>

// ---------------------------------------------------------------------------
// MI455X (gfx1250) fused attention block, bf16 WMMA + async-to-LDS staging.
//   B=2, T=2048, D=2048, H=16, d=128.
// ---------------------------------------------------------------------------

typedef __attribute__((ext_vector_type(16))) __bf16 v16bf;
typedef __attribute__((ext_vector_type(8)))  float  v8f;

union BFrag { v16bf v; unsigned u[8]; };
union AccF  { v8f   v; float    f[8]; };

#define EPSF 1.1920929e-07f   // np.finfo(np.float32).eps

__device__ __forceinline__ unsigned short f2bf(float f) {
  unsigned x = __float_as_uint(f);
  x += 0x7fffu + ((x >> 16) & 1u);          // round-to-nearest-even
  return (unsigned short)(x >> 16);
}

__device__ __forceinline__ v8f wmma_bf16(v16bf a, v16bf b, v8f c) {
  // v_wmma_f32_16x16x32_bf16  D = A(16x32) * B(32x16) + C(16x16 f32)
  return __builtin_amdgcn_wmma_f32_16x16x32_bf16(
      false, a, false, b, (short)0, c, false, false);
}

// Async global -> LDS copy of 16 bytes per lane (CDNA5 GLOBAL_LOAD_ASYNC_TO_LDS,
// tracked by ASYNCcnt; bypasses VGPRs). lptr low 32 bits = LDS byte offset.
__device__ __forceinline__ void async_copy_b128(const void* gptr, void* lptr) {
  unsigned loff = (unsigned)(unsigned long long)lptr;
  asm volatile("global_load_async_to_lds_b128 %0, %1, off"
               :: "v"(loff), "v"(gptr) : "memory");
}
__device__ __forceinline__ void async_wait0() {
  asm volatile("s_wait_asynccnt 0x0" ::: "memory");
}

// ---------------------------------------------------------------------------
// Kernel 1: row RMSNorm of x -> two bf16 tensors (g_q and g_kv gains).
// One wave per row of 2048; 64 elements per lane; shuffle reduction.
// ---------------------------------------------------------------------------
__global__ void __launch_bounds__(256)
k_rmsnorm_x(const float* __restrict__ x,
            const float* __restrict__ g_q,
            const float* __restrict__ g_kv,
            unsigned short* __restrict__ xq,
            unsigned short* __restrict__ xkv) {
  const int wave = threadIdx.x >> 5, lane = threadIdx.x & 31;
  const int row  = blockIdx.x * 8 + wave;           // 0..4095
  const float* xr = x + (size_t)row * 2048;
  float vals[64];
  float ss = 0.0f;
#pragma unroll
  for (int i = 0; i < 64; ++i) {
    float v = xr[lane + i * 32];
    vals[i] = v;
    ss += v * v;
  }
#pragma unroll
  for (int m = 16; m; m >>= 1) ss += __shfl_xor(ss, m);
  const float inv = rsqrtf(ss * (1.0f / 2048.0f) + EPSF);
  unsigned short* oq = xq  + (size_t)row * 2048;
  unsigned short* ok = xkv + (size_t)row * 2048;
#pragma unroll
  for (int i = 0; i < 64; ++i) {
    int j = lane + i * 32;
    float v = vals[i] * inv;
    oq[j] = f2bf(v * g_q[j]);
    ok[j] = f2bf(v * g_kv[j]);
  }
}

// ---------------------------------------------------------------------------
// Kernel 2: fp32 -> bf16 cast (weights), grid-stride.
// ---------------------------------------------------------------------------
__global__ void __launch_bounds__(256)
k_cast_bf16(const float* __restrict__ in, unsigned short* __restrict__ out, int n) {
  int i = blockIdx.x * blockDim.x + threadIdx.x;
  int stride = gridDim.x * blockDim.x;
  for (; i < n; i += stride) out[i] = f2bf(in[i]);
}

// ---------------------------------------------------------------------------
// Kernel 3: generic bf16 GEMM, fp32 accumulate + bias.
//   C[M,N] = A[M,K] @ B[K,N] + bias[N]
// Block: 256 threads (8 waves), tile 128x64, K-step 32.
// A staged via async-to-LDS; B staged transposed (manual) so B-fragment
// K-pairs are contiguous 32-bit LDS loads.
// ---------------------------------------------------------------------------
#define G_BM 128
#define G_BN 64
#define G_BK 32
#define LSTR 40    // padded LDS row stride (bf16 elements)

__global__ void __launch_bounds__(256)
k_gemm_bf16(const unsigned short* __restrict__ A,
            const unsigned short* __restrict__ B,
            const float* __restrict__ bias,
            float* __restrict__ C,
            int M, int N, int K) {
  __shared__ unsigned short lA[G_BM * LSTR];
  __shared__ unsigned short lBt[G_BN * LSTR];

  const int tid  = threadIdx.x;
  const int lane = tid & 31, wave = tid >> 5;
  const int wr   = wave >> 1, wc = wave & 1;          // 4x2 wave grid
  const int half = lane >> 4, l16 = lane & 15;

  const int block_m = blockIdx.y * G_BM;
  const int block_n = blockIdx.x * G_BN;

  AccF acc[2][2];
#pragma unroll
  for (int i = 0; i < 2; ++i)
#pragma unroll
    for (int j = 0; j < 2; ++j)
#pragma unroll
      for (int r = 0; r < 8; ++r) acc[i][j].f[r] = 0.0f;

  for (int k0 = 0; k0 < K; k0 += G_BK) {
    // ---- stage A tile 128x32 via async-to-LDS (2 threads/row, 16 bf16 each)
    {
      int row = tid >> 1;
      int col = (tid & 1) * 16;
      const unsigned short* src = A + (size_t)(block_m + row) * K + k0 + col;
      unsigned short* dst = lA + row * LSTR + col;
      async_copy_b128(src, dst);
      async_copy_b128(src + 8, dst + 8);
    }
    // ---- stage B tile 32x64 transposed -> lBt[n][k]
    {
      int kk = tid >> 3;              // 0..31
      int ng = (tid & 7) * 8;         // 0..56
      uint4 d = *(const uint4*)(B + (size_t)(k0 + kk) * N + block_n + ng);
      unsigned short e[8];
      *(uint4*)e = d;
#pragma unroll
      for (int i = 0; i < 8; ++i) lBt[(ng + i) * LSTR + kk] = e[i];
    }
    async_wait0();
    __syncthreads();

    BFrag a[2], b[2];
#pragma unroll
    for (int ti = 0; ti < 2; ++ti) {
      int m = wr * 32 + ti * 16 + l16;
#pragma unroll
      for (int r = 0; r < 8; ++r) {
        int k = ((r >> 2) << 4) + (half << 3) + ((r & 3) << 1);
        a[ti].u[r] = *(const unsigned int*)(lA + m * LSTR + k);
      }
    }
#pragma unroll
    for (int tj = 0; tj < 2; ++tj) {
      int n = wc * 32 + tj * 16 + l16;
#pragma unroll
      for (int r = 0; r < 8; ++r) {
        int k = ((r >> 2) << 4) + (half << 3) + ((r & 3) << 1);
        b[tj].u[r] = *(const unsigned int*)(lBt + n * LSTR + k);
      }
    }
#pragma unroll
    for (int ti = 0; ti < 2; ++ti)
#pragma unroll
      for (int tj = 0; tj < 2; ++tj)
        acc[ti][tj].v = wmma_bf16(a[ti].v, b[tj].v, acc[ti][tj].v);
    __syncthreads();
  }

  // ---- epilogue: C = acc + bias
#pragma unroll
  for (int ti = 0; ti < 2; ++ti)
#pragma unroll
    for (int tj = 0; tj < 2; ++tj)
#pragma unroll
      for (int r = 0; r < 8; ++r) {
        int m = block_m + wr * 32 + ti * 16 + r + 8 * half;
        int n = block_n + wc * 32 + tj * 16 + l16;
        C[(size_t)m * N + n] = acc[ti][tj].f[r] + bias[n];
      }
}

// ---------------------------------------------------------------------------
// Kernel 4: per-head RMSNorm of q and k (+fold 1/sqrt(d) scale into q),
// bf16 cast, layout to [B,H,T,d]; v cast + transpose to [B,H,d,T].
// One wave per (b,t,h); 4 elements per lane.
// ---------------------------------------------------------------------------
__global__ void __launch_bounds__(256)
k_head_stage(const float* __restrict__ qf,    // [4096, 2048]
             const float* __restrict__ kvf,   // [4096, 4096]
             const float* __restrict__ g_head,
             unsigned short* __restrict__ qn, // [B*H, T, 128]
             unsigned short* __restrict__ kn, // [B*H, T, 128]
             unsigned short* __restrict__ vt) // [B*H, 128, T]
{
  const int wave = threadIdx.x >> 5, lane = threadIdx.x & 31;
  const int idx = blockIdx.x * 8 + wave;     // 0..65535 = (row*16 + h)
  const int h   = idx & 15;
  const int row = idx >> 4;                  // b*2048 + t
  const int b   = row >> 11, t = row & 2047;
  const float qscale = 0.08838834764831845f; // d^-0.5, d = 128

  const size_t obase = ((size_t)(b * 16 + h) * 2048 + t) * 128;

  // q branch
  {
    const float* src = qf + (size_t)row * 2048 + h * 128;
    float v[4]; float ss = 0.0f;
#pragma unroll
    for (int i = 0; i < 4; ++i) { v[i] = src[lane + i * 32]; ss += v[i] * v[i]; }
#pragma unroll
    for (int m = 16; m; m >>= 1) ss += __shfl_xor(ss, m);
    float inv = rsqrtf(ss * (1.0f / 128.0f) + EPSF);
#pragma unroll
    for (int i = 0; i < 4; ++i) {
      int j = lane + i * 32;
      qn[obase + j] = f2bf(v[i] * inv * g_head[j] * qscale);
    }
  }
  // k branch
  {
    const float* src = kvf + (size_t)row * 4096 + h * 128;
    float v[4]; float ss = 0.0f;
#pragma unroll
    for (int i = 0; i < 4; ++i) { v[i] = src[lane + i * 32]; ss += v[i] * v[i]; }
#pragma unroll
    for (int m = 16; m; m >>= 1) ss += __shfl_xor(ss, m);
    float inv = rsqrtf(ss * (1.0f / 128.0f) + EPSF);
#pragma unroll
    for (int i = 0; i < 4; ++i) {
      int j = lane + i * 32;
      kn[obase + j] = f2bf(v[i] * inv * g_head[j]);
    }
  }
  // v branch (no norm), transposed store
  {
    const float* src = kvf + (size_t)row * 4096 + 2048 + h * 128;
    const size_t vb = (size_t)(b * 16 + h) * 128 * 2048;
#pragma unroll
    for (int i = 0; i < 4; ++i) {
      int j = lane + i * 32;
      vt[vb + (size_t)j * 2048 + t] = f2bf(src[j]);
    }
  }
}

// ---------------------------------------------------------------------------
// Kernel 5: flash attention. Grid (T/128, B*H). 8 waves/block, each wave
// owns a 16-query strip; 32-key tiles; online softmax; O accumulated fp32.
// K/V tiles staged once per block via async-to-LDS (8x traffic reduction
// vs. per-wave global fragment loads). Output written bf16 directly in
// [B,T,H*d] layout for the out-proj GEMM.
// ---------------------------------------------------------------------------
#define KSTR 136   // lK row stride (bf16): 16 lanes * 272B -> conflict-free
#define VSTR 40    // lV row stride (bf16)

__global__ void __launch_bounds__(256)
k_flash_attn(const unsigned short* __restrict__ qn,  // [B*H, T, 128] (pre-scaled)
             const unsigned short* __restrict__ kn,  // [B*H, T, 128]
             const unsigned short* __restrict__ vt,  // [B*H, 128, T]
             unsigned short* __restrict__ wv)        // [B*T, 2048]
{
  __shared__ unsigned short lP[8 * 16 * LSTR];       // per-wave P staging
  __shared__ unsigned short lK[32 * KSTR];           // K tile  [key][d]
  __shared__ unsigned short lV[128 * VSTR];          // V tile  [d][key]

  const int tid = threadIdx.x;
  const int wave = tid >> 5, lane = tid & 31;
  const int half = lane >> 4, l16 = lane & 15;
  const int bh = blockIdx.y;                          // b*16 + h
  const int q0 = blockIdx.x * 128 + wave * 16;

  const unsigned short* Q  = qn + (size_t)bh * 2048 * 128;
  const unsigned short* Kp = kn + (size_t)bh * 2048 * 128;
  const unsigned short* Vt = vt + (size_t)bh * 128 * 2048;
  unsigned short* pbuf = lP + wave * 16 * LSTR;

  // persistent Q fragments (16 x 128 = 4 chunks of 16x32)
  BFrag qa[4];
  {
    const int qm = q0 + l16;
#pragma unroll
    for (int c = 0; c < 4; ++c)
#pragma unroll
      for (int r = 0; r < 8; ++r) {
        int k = c * 32 + ((r >> 2) << 4) + (half << 3) + ((r & 3) << 1);
        qa[c].u[r] = *(const unsigned int*)(Q + (size_t)qm * 128 + k);
      }
  }

  AccF o[8];
#pragma unroll
  for (int cn = 0; cn < 8; ++cn)
#pragma unroll
    for (int r = 0; r < 8; ++r) o[cn].f[r] = 0.0f;
  float m_run[8], l_run[8];
#pragma unroll
  for (int r = 0; r < 8; ++r) { m_run[r] = -1e30f; l_run[r] = 0.0f; }

  // staging coordinates (fixed per thread)
  const int krow = tid >> 3;              // 0..31  (key within tile)
  const int kcol = (tid & 7) * 16;        // 0..112 (d offset)
  const int vrow = tid >> 1;              // 0..127 (d index)
  const int vcol = (tid & 1) * 16;        // 0/16   (key offset)

  for (int j0 = 0; j0 < 2048; j0 += 32) {
    // ---- stage K (32x128) and V (128x32) tiles via async-to-LDS
    {
      const unsigned short* ks = Kp + (size_t)(j0 + krow) * 128 + kcol;
      unsigned short* kd = lK + krow * KSTR + kcol;
      async_copy_b128(ks, kd);
      async_copy_b128(ks + 8, kd + 8);
      const unsigned short* vs = Vt + (size_t)vrow * 2048 + j0 + vcol;
      unsigned short* vd = lV + vrow * VSTR + vcol;
      async_copy_b128(vs, vd);
      async_copy_b128(vs + 8, vd + 8);
    }
    async_wait0();
    __syncthreads();                       // (a) tiles visible to all waves

    // ---- S = Q @ K^T for 32 keys (two 16-key column tiles)
    AccF s[2];
#pragma unroll
    for (int nt = 0; nt < 2; ++nt) {
#pragma unroll
      for (int r = 0; r < 8; ++r) s[nt].f[r] = 0.0f;
      const int n = nt * 16 + l16;         // key within tile
#pragma unroll
      for (int c = 0; c < 4; ++c) {
        BFrag kb;
#pragma unroll
        for (int r = 0; r < 8; ++r) {
          int kk = c * 32 + ((r >> 2) << 4) + (half << 3) + ((r & 3) << 1);   // d index
          kb.u[r] = *(const unsigned int*)(lK + n * KSTR + kk);
        }
        s[nt].v = wmma_bf16(qa[c].v, kb.v, s[nt].v);
      }
    }

    // ---- online softmax stats (rows live across the 16-lane groups)
    float alpha[8];
#pragma unroll
    for (int r = 0; r < 8; ++r) {
      float v = fmaxf(s[0].f[r], s[1].f[r]);
#pragma unroll
      for (int m = 8; m; m >>= 1) v = fmaxf(v, __shfl_xor(v, m));
      float mn = fmaxf(m_run[r], v);
      alpha[r] = __expf(m_run[r] - mn);
      m_run[r] = mn;
    }
#pragma unroll
    for (int r = 0; r < 8; ++r) {
      float p0 = __expf(s[0].f[r] - m_run[r]);
      float p1 = __expf(s[1].f[r] - m_run[r]);
      s[0].f[r] = p0; s[1].f[r] = p1;
      float v = p0 + p1;
#pragma unroll
      for (int m = 8; m; m >>= 1) v += __shfl_xor(v, m);
      l_run[r] = l_run[r] * alpha[r] + v;
    }
#pragma unroll
    for (int cn = 0; cn < 8; ++cn)
#pragma unroll
      for (int r = 0; r < 8; ++r) o[cn].f[r] *= alpha[r];

    // ---- C-layout P -> A-layout bf16 via LDS
#pragma unroll
    for (int nt = 0; nt < 2; ++nt)
#pragma unroll
      for (int r = 0; r < 8; ++r) {
        int m = r + 8 * half;
        int n = nt * 16 + l16;
        pbuf[m * LSTR + n] = f2bf(s[nt].f[r]);
      }
    __syncthreads();                       // (b) P write -> read ordering
    BFrag pa;
#pragma unroll
    for (int r = 0; r < 8; ++r) {
      int k = ((r >> 2) << 4) + (half << 3) + ((r & 3) << 1);
      pa.u[r] = *(const unsigned int*)(pbuf + l16 * LSTR + k);
    }

    // ---- O += P @ V  (8 column chunks of d)
#pragma unroll
    for (int cn = 0; cn < 8; ++cn) {
      BFrag vb;
      const int n = cn * 16 + l16;                    // d column
#pragma unroll
      for (int r = 0; r < 8; ++r) {
        int kk = ((r >> 2) << 4) + (half << 3) + ((r & 3) << 1);              // key within tile
        vb.u[r] = *(const unsigned int*)(lV + n * VSTR + kk);
      }
      o[cn].v = wmma_bf16(pa.v, vb.v, o[cn].v);
    }
    __syncthreads();                       // (c) done reading lK/lV/pbuf
  }

  // ---- normalize + store to [B,T, H*d] (bf16 for out-proj GEMM)
  const int b = bh >> 4, h = bh & 15;
  float invl[8];
#pragma unroll
  for (int r = 0; r < 8; ++r) invl[r] = 1.0f / l_run[r];
#pragma unroll
  for (int cn = 0; cn < 8; ++cn)
#pragma unroll
    for (int r = 0; r < 8; ++r) {
      int mrow = q0 + r + 8 * half;
      int col  = h * 128 + cn * 16 + l16;
      wv[((size_t)(b * 2048 + mrow)) * 2048 + col] = f2bf(o[cn].f[r] * invl[r]);
    }
}

// ---------------------------------------------------------------------------
// Host orchestration
// ---------------------------------------------------------------------------
extern "C" void kernel_launch(void* const* d_in, const int* in_sizes, int n_in,
                              void* d_out, int out_size, void* d_ws, size_t ws_size,
                              hipStream_t stream) {
  (void)in_sizes; (void)n_in; (void)out_size; (void)ws_size;
  const float* x      = (const float*)d_in[0];
  const float* g_q    = (const float*)d_in[1];
  const float* Wq     = (const float*)d_in[2];
  const float* bq     = (const float*)d_in[3];
  const float* g_kv   = (const float*)d_in[4];
  const float* Wkv    = (const float*)d_in[5];
  const float* bkv    = (const float*)d_in[6];
  const float* g_head = (const float*)d_in[7];
  const float* Wo     = (const float*)d_in[8];
  const float* bo     = (const float*)d_in[9];
  float* out = (float*)d_out;

  const int M = 4096, D = 2048;

  // workspace carve (~225 MB)
  size_t off = 0;
  auto carve = [&](size_t bytes) -> void* {
    void* p = (char*)d_ws + off;
    off += (bytes + 255) & ~(size_t)255;
    return p;
  };
  unsigned short* xq     = (unsigned short*)carve((size_t)M * D * 2);
  unsigned short* xkv    = (unsigned short*)carve((size_t)M * D * 2);
  unsigned short* Wq_b   = (unsigned short*)carve((size_t)D * D * 2);
  unsigned short* Wkv_b  = (unsigned short*)carve((size_t)D * 2 * D * 2);
  unsigned short* Wo_b   = (unsigned short*)carve((size_t)D * D * 2);
  float*          q_f32  = (float*)carve((size_t)M * D * 4);
  float*          kv_f32 = (float*)carve((size_t)M * 2 * D * 4);
  unsigned short* qn     = (unsigned short*)carve((size_t)M * D * 2);
  unsigned short* kn     = (unsigned short*)carve((size_t)M * D * 2);
  unsigned short* vt     = (unsigned short*)carve((size_t)M * D * 2);
  unsigned short* wv     = (unsigned short*)carve((size_t)M * D * 2);

  // 1) RMSNorm(x) -> bf16 (both gains)
  k_rmsnorm_x<<<512, 256, 0, stream>>>(x, g_q, g_kv, xq, xkv);
  // 2) weight casts
  k_cast_bf16<<<4096, 256, 0, stream>>>(Wq,  Wq_b,  D * D);
  k_cast_bf16<<<4096, 256, 0, stream>>>(Wkv, Wkv_b, D * 2 * D);
  k_cast_bf16<<<4096, 256, 0, stream>>>(Wo,  Wo_b,  D * D);
  // 3) Q and KV projections (fp32 accum)
  k_gemm_bf16<<<dim3(D / G_BN, M / G_BM), 256, 0, stream>>>(
      xq, Wq_b, bq, q_f32, M, D, D);
  k_gemm_bf16<<<dim3(2 * D / G_BN, M / G_BM), 256, 0, stream>>>(
      xkv, Wkv_b, bkv, kv_f32, M, 2 * D, D);
  // 4) per-head RMSNorm + layout
  k_head_stage<<<8192, 256, 0, stream>>>(q_f32, kv_f32, g_head, qn, kn, vt);
  // 5) flash attention
  k_flash_attn<<<dim3(16, 32), 256, 0, stream>>>(qn, kn, vt, wv);
  // 6) output projection -> fp32 d_out
  k_gemm_bf16<<<dim3(D / G_BN, M / G_BM), 256, 0, stream>>>(
      wv, Wo_b, bo, out, M, D, D);
}
